// GlobalContextAwareAttention_6476810682622
// MI455X (gfx1250) — matrix operand order repaired and checked by reading, not compiled, not run on hardware
//
#include <hip/hip_runtime.h>
#include <hip/hip_bf16.h>
#include <math.h>

// ---------- types ----------
typedef __attribute__((ext_vector_type(16))) __bf16 v16bf;
typedef __attribute__((ext_vector_type(8)))  __bf16 v8bf;
typedef __attribute__((ext_vector_type(8)))  float  v8f;

#define S_LEN 2048
#define BATCH 2
#define DMODEL 1024
#define NHEAD 16
#define DK 64
#define NROWS (S_LEN * BATCH)   // 4096

__device__ __forceinline__ unsigned short f32_to_bf16_bits(float f) {
  union { float f; unsigned u; } v; v.f = f;
  unsigned u = v.u + 0x7FFFu + ((v.u >> 16) & 1u);
  return (unsigned short)(u >> 16);
}
__device__ __forceinline__ __bf16 f32_to_bf16(float f) {
  union { unsigned short s; __bf16 h; } c; c.s = f32_to_bf16_bits(f);
  return c.h;
}
__device__ __forceinline__ unsigned pack_bf16x2(float lo, float hi) {
  return (unsigned)f32_to_bf16_bits(lo) | ((unsigned)f32_to_bf16_bits(hi) << 16);
}
__device__ __forceinline__ v16bf cat8(v8bf lo, v8bf hi) {
  return __builtin_shufflevector(lo, hi, 0,1,2,3,4,5,6,7,8,9,10,11,12,13,14,15);
}

// =====================================================================
// Elementwise f32 -> bf16 (8 elems / thread, 16B stores)
// =====================================================================
__global__ __launch_bounds__(256)
void convert_bf16_kernel(const float* __restrict__ X, __bf16* __restrict__ Y) {
  const size_t i = ((size_t)blockIdx.x * 256 + threadIdx.x) * 8;
  float4 a = *(const float4*)(X + i);
  float4 b = *(const float4*)(X + i + 4);
  uint4 o;
  o.x = pack_bf16x2(a.x, a.y);
  o.y = pack_bf16x2(a.z, a.w);
  o.z = pack_bf16x2(b.x, b.y);
  o.w = pack_bf16x2(b.z, b.w);
  *(uint4*)(Y + i) = o;
}

// =====================================================================
// Transpose + convert: Wt[n][k] = bf16(W[k][n]);  K == N == 1024.
// 32x32 tiles through LDS (33-float rows to dodge bank conflicts).
// =====================================================================
__global__ __launch_bounds__(256)
void transpose_convert_kernel(const float* __restrict__ W, __bf16* __restrict__ Wt) {
  __shared__ float tile[32][33];
  const int kBase = blockIdx.x * 32;
  const int nBase = blockIdx.y * 32;
  const int tx  = threadIdx.x & 31;
  const int ty4 = (threadIdx.x >> 5) * 4;
  for (int r = 0; r < 4; ++r)
    tile[ty4 + r][tx] = W[(size_t)(kBase + ty4 + r) * DMODEL + nBase + tx];
  __syncthreads();
  for (int r = 0; r < 4; ++r)
    Wt[(size_t)(nBase + ty4 + r) * DMODEL + kBase + tx] =
        f32_to_bf16(tile[tx][ty4 + r]);
}

// =====================================================================
// Tiled bf16 GEMM with bias: C[M,N] = A[M,K] * Bt[N,K]^T + bias[N]
// A bf16 row-major [m][k]; Bt bf16 row-major [n][k] (pre-transposed).
// BM=BN=128, BK=32. 256 threads = 8 waves; wave tile 32x64 = 2x4 WMMA.
// Register double-buffer: load tile k+1 while computing tile k from LDS.
// =====================================================================
__global__ __launch_bounds__(256)
void gemm_bf16_kernel(const __bf16* __restrict__ A, const __bf16* __restrict__ Bt,
                      const float* __restrict__ bias, float* __restrict__ C,
                      int M, int N, int K) {
  __shared__ __align__(16) __bf16 As[128 * 32];     // [m][k]
  __shared__ __align__(16) __bf16 Bs[128 * 32];     // [n][k]

  const int tid  = threadIdx.x;
  const int wid  = tid >> 5;
  const int lane = tid & 31;
  const int g    = lane >> 4;
  const int lm   = lane & 15;
  const int mBlock = blockIdx.x * 128;
  const int nBlock = blockIdx.y * 128;
  const int waveM  = (wid >> 1) * 32;
  const int waveN  = (wid & 1) * 64;

  v8f acc[2][4];
  for (int i = 0; i < 2; ++i)
    for (int j = 0; j < 4; ++j)
      for (int e = 0; e < 8; ++e) acc[i][j][e] = 0.0f;

  // Each tile = 128 rows x 32 bf16 = 512 x 16B chunks; 2 chunks per thread.
  uint4 ra[2], rb[2];
  auto loadTiles = [&](int k0) {
    for (int r = 0; r < 2; ++r) {
      int idx = r * 256 + tid;
      int row = idx >> 2;
      int c8  = (idx & 3) * 8;
      ra[r] = *(const uint4*)(A  + (size_t)(mBlock + row) * K + k0 + c8);
      rb[r] = *(const uint4*)(Bt + (size_t)(nBlock + row) * K + k0 + c8);
    }
  };
  loadTiles(0);

  for (int k0 = 0; k0 < K; k0 += 32) {
    // commit current regs to LDS
    for (int r = 0; r < 2; ++r) {
      int idx = r * 256 + tid;
      int row = idx >> 2;
      int c8  = (idx & 3) * 8;
      *(uint4*)&As[row * 32 + c8] = ra[r];
      *(uint4*)&Bs[row * 32 + c8] = rb[r];
    }
    __syncthreads();

    // prefetch tile k+2, fetch tile k+1 into regs
    if (k0 + 64 < K) {
      int row = tid & 127;
      __builtin_prefetch(A  + (size_t)(mBlock + row) * K + k0 + 64, 0, 1);
      __builtin_prefetch(Bt + (size_t)(nBlock + row) * K + k0 + 64, 0, 1);
    }
    if (k0 + 32 < K) loadTiles(k0 + 32);

    // A fragments (16x32; lane = row, 16 contiguous K)
    v16bf afrag[2];
    for (int i = 0; i < 2; ++i) {
      const __bf16* base = &As[(waveM + i * 16 + lm) * 32];
      v8bf lo = *(const v8bf*)(base + g * 16);
      v8bf hi = *(const v8bf*)(base + g * 16 + 8);
      afrag[i] = cat8(lo, hi);
    }
    // B fragments (32x16; lane = column, 16 contiguous K)
    v16bf bfrag[4];
    for (int j = 0; j < 4; ++j)
      bfrag[j] = *(const v16bf*)&Bs[(waveN + j * 16 + lm) * 32 + g * 16];

    for (int i = 0; i < 2; ++i)
      for (int j = 0; j < 4; ++j)
        acc[i][j] = __builtin_amdgcn_wmma_f32_16x16x32_bf16(
            false, afrag[i], false, bfrag[j], (short)0, acc[i][j], false, false);
    __syncthreads();
  }

  // epilogue: C layout lane l -> (m = r + 8*(l>>4), n = l&15)
  for (int i = 0; i < 2; ++i)
    for (int j = 0; j < 4; ++j) {
      int col = nBlock + waveN + j * 16 + lm;
      float bv = bias[col];
      for (int r = 0; r < 8; ++r) {
        int row = mBlock + waveM + i * 16 + r + 8 * g;
        C[(size_t)row * N + col] = acc[i][j][r] + bv;
      }
    }
}

// =====================================================================
// Gate + mix + pack kernel. One block per row m (m = s*B + b).
// lam_q = sigmoid(qx_p.Wvqx + bqx + qc_p.Wvqc + bqc); same for k.
// q,k -> bf16 [B,H,S,DK]; v -> bf16 transposed [B,H,DK,S].
// =====================================================================
__global__ __launch_bounds__(256)
void gate_pack_kernel(const float* __restrict__ qx_p, const float* __restrict__ qc_p,
                      const float* __restrict__ kx_p, const float* __restrict__ kc_p,
                      const float* __restrict__ vx_p,
                      const float* __restrict__ Wvqx, const float* __restrict__ bvqx,
                      const float* __restrict__ Wvqc, const float* __restrict__ bvqc,
                      const float* __restrict__ Wvkx, const float* __restrict__ bvkx,
                      const float* __restrict__ Wvkc, const float* __restrict__ bvkc,
                      __bf16* __restrict__ qb, __bf16* __restrict__ kb,
                      __bf16* __restrict__ vT) {
  const int m   = blockIdx.x;
  const int tid = threadIdx.x;
  const size_t base = (size_t)m * DMODEL;

  float xq[4], cq[4], xk[4], ck[4], xv[4];
  float pqx = 0.f, pqc = 0.f, pkx = 0.f, pkc = 0.f;
  for (int e = 0; e < 4; ++e) {
    int d = tid * 4 + e;
    xq[e] = qx_p[base + d]; cq[e] = qc_p[base + d];
    xk[e] = kx_p[base + d]; ck[e] = kc_p[base + d];
    xv[e] = vx_p[base + d];
    pqx += xq[e] * Wvqx[d];
    pqc += cq[e] * Wvqc[d];
    pkx += xk[e] * Wvkx[d];
    pkc += ck[e] * Wvkc[d];
  }
  for (int off = 16; off >= 1; off >>= 1) {
    pqx += __shfl_xor(pqx, off, 32);
    pqc += __shfl_xor(pqc, off, 32);
    pkx += __shfl_xor(pkx, off, 32);
    pkc += __shfl_xor(pkc, off, 32);
  }
  __shared__ float sred[8][4];
  __shared__ float lam[2];
  if ((tid & 31) == 0) {
    sred[tid >> 5][0] = pqx; sred[tid >> 5][1] = pqc;
    sred[tid >> 5][2] = pkx; sred[tid >> 5][3] = pkc;
  }
  __syncthreads();
  if (tid == 0) {
    float sq = 0.f, sc = 0.f, skx = 0.f, skc = 0.f;
    for (int w = 0; w < 8; ++w) {
      sq += sred[w][0]; sc += sred[w][1]; skx += sred[w][2]; skc += sred[w][3];
    }
    float zq = sq + bvqx[0] + sc + bvqc[0];
    float zk = skx + bvkx[0] + skc + bvkc[0];
    lam[0] = 1.f / (1.f + __expf(-zq));
    lam[1] = 1.f / (1.f + __expf(-zk));
  }
  __syncthreads();
  const float lq = lam[0], lk = lam[1];
  const int s = m >> 1;                   // B == 2
  const int b = m & 1;
  for (int e = 0; e < 4; ++e) {
    int d  = tid * 4 + e;
    int h  = d >> 6;
    int dk = d & 63;
    size_t qoff = ((size_t)(b * NHEAD + h) * S_LEN + s) * DK + dk;
    qb[qoff] = f32_to_bf16((1.f - lq) * xq[e] + lq * cq[e]);
    kb[qoff] = f32_to_bf16((1.f - lk) * xk[e] + lk * ck[e]);
    size_t voff = ((size_t)(b * NHEAD + h) * DK + dk) * S_LEN + s;
    vT[voff] = f32_to_bf16(xv[e]);
  }
}

// =====================================================================
// Flash attention. blockIdx.y = b*H + h, blockIdx.x = q block of 128.
// 8 waves x 16 q rows, all 2048 keys in 64-key chunks.
// QK^T and P*V via v_wmma_f32_16x16x32_bf16; online softmax via shfl_xor.
// P f32 (C layout) -> LDS bf16 -> A-layout fragments (s_wait_dscnt ordered).
// =====================================================================
__global__ __launch_bounds__(256)
void flash_attn_kernel(const __bf16* __restrict__ Q, const __bf16* __restrict__ Kmat,
                       const __bf16* __restrict__ VT, __bf16* __restrict__ ctx) {
  const int tid  = threadIdx.x;
  const int wid  = tid >> 5;
  const int lane = tid & 31;
  const int g    = lane >> 4;
  const int lm   = lane & 15;
  const int bh   = blockIdx.y;
  const int b    = bh >> 4;              // H == 16
  const int h    = bh & 15;
  const int qBase = blockIdx.x * 128 + wid * 16;

  const __bf16* Qh = Q    + (size_t)bh * S_LEN * DK;
  const __bf16* Kh = Kmat + (size_t)bh * S_LEN * DK;
  const __bf16* Vh = VT   + (size_t)bh * DK * S_LEN;

  __shared__ __align__(16) __bf16 Pl[8][16 * 64];

  v16bf aq[2];
  for (int kc = 0; kc < 2; ++kc) {
    const __bf16* base = Qh + (size_t)(qBase + lm) * DK + kc * 32;
    v8bf lo = *(const v8bf*)(base + g * 16);
    v8bf hi = *(const v8bf*)(base + g * 16 + 8);
    aq[kc] = cat8(lo, hi);
  }

  v8f cacc[4];
  for (int j = 0; j < 4; ++j)
    for (int e = 0; e < 8; ++e) cacc[j][e] = 0.0f;
  float run_max[8], run_sum[8];
  for (int r = 0; r < 8; ++r) { run_max[r] = -INFINITY; run_sum[r] = 0.f; }

  const float scale = 0.125f;  // 1/sqrt(64)

  for (int k0 = 0; k0 < S_LEN; k0 += 64) {
    v8f sacc[4];
    for (int j = 0; j < 4; ++j)
      for (int e = 0; e < 8; ++e) sacc[j][e] = 0.0f;
    for (int j = 0; j < 4; ++j)
      for (int kc = 0; kc < 2; ++kc) {
        const __bf16* base = Kh + (size_t)(k0 + j * 16 + lm) * DK + kc * 32 + g * 16;
        v16bf bfrag = *(const v16bf*)base;
        sacc[j] = __builtin_amdgcn_wmma_f32_16x16x32_bf16(
            false, aq[kc], false, bfrag, (short)0, sacc[j], false, false);
      }

    float nmax[8];
    for (int r = 0; r < 8; ++r) {
      float mx = -INFINITY;
      for (int j = 0; j < 4; ++j) mx = fmaxf(mx, sacc[j][r] * scale);
      for (int off = 1; off < 16; off <<= 1) mx = fmaxf(mx, __shfl_xor(mx, off, 16));
      nmax[r] = fmaxf(run_max[r], mx);
    }
    for (int r = 0; r < 8; ++r) {
      float factor = __expf(run_max[r] - nmax[r]);
      run_sum[r] *= factor;
      for (int j = 0; j < 4; ++j) cacc[j][r] *= factor;
      run_max[r] = nmax[r];
    }
    float psum[8];
    for (int r = 0; r < 8; ++r) psum[r] = 0.f;
    for (int j = 0; j < 4; ++j)
      for (int r = 0; r < 8; ++r) {
        float p = __expf(sacc[j][r] * scale - nmax[r]);
        psum[r] += p;
        Pl[wid][(r + 8 * g) * 64 + j * 16 + lm] = f32_to_bf16(p);
      }
    for (int r = 0; r < 8; ++r) {
      float sm = psum[r];
      for (int off = 1; off < 16; off <<= 1) sm += __shfl_xor(sm, off, 16);
      run_sum[r] += sm;
    }
    asm volatile("s_wait_dscnt 0" ::: "memory");

    for (int kc = 0; kc < 2; ++kc) {
      const __bf16* pb = &Pl[wid][lm * 64 + kc * 32];
      v8bf lo = *(const v8bf*)(pb + g * 16);
      v8bf hi = *(const v8bf*)(pb + g * 16 + 8);
      v16bf ap = cat8(lo, hi);
      for (int j = 0; j < 4; ++j) {
        const __bf16* vb = Vh + (size_t)(j * 16 + lm) * S_LEN + k0 + kc * 32 + g * 16;
        v16bf bv = *(const v16bf*)vb;
        cacc[j] = __builtin_amdgcn_wmma_f32_16x16x32_bf16(
            false, ap, false, bv, (short)0, cacc[j], false, false);
      }
    }
  }

  for (int j = 0; j < 4; ++j)
    for (int r = 0; r < 8; ++r) {
      int sq = qBase + r + 8 * g;
      int dk = h * DK + j * 16 + lm;
      float val = cacc[j][r] / run_sum[r];
      ctx[((size_t)sq * BATCH + b) * DMODEL + dk] = f32_to_bf16(val);
    }
}

// =====================================================================
// Host launch
// =====================================================================
extern "C" void kernel_launch(void* const* d_in, const int* in_sizes, int n_in,
                              void* d_out, int out_size, void* d_ws, size_t ws_size,
                              hipStream_t stream) {
  (void)in_sizes; (void)n_in; (void)out_size; (void)ws_size;

  const float* act[5] = { (const float*)d_in[0], (const float*)d_in[1],
                          (const float*)d_in[2], (const float*)d_in[3],
                          (const float*)d_in[4] };           // qx kx vx qc kc
  const float* Wlin[6]  = { (const float*)d_in[5],  (const float*)d_in[7],
                            (const float*)d_in[9],  (const float*)d_in[11],
                            (const float*)d_in[13], (const float*)d_in[15] };
  const float* blin[6]  = { (const float*)d_in[6],  (const float*)d_in[8],
                            (const float*)d_in[10], (const float*)d_in[12],
                            (const float*)d_in[14], (const float*)d_in[16] };
  const float* W_Vqx = (const float*)d_in[17]; const float* b_Vqx = (const float*)d_in[18];
  const float* W_Vqc = (const float*)d_in[19]; const float* b_Vqc = (const float*)d_in[20];
  const float* W_Vkx = (const float*)d_in[21]; const float* b_Vkx = (const float*)d_in[22];
  const float* W_Vkc = (const float*)d_in[23]; const float* b_Vkc = (const float*)d_in[24];

  // ---- workspace layout (with aliasing; peak ~132 MiB) ----
  const size_t actBf   = (size_t)NROWS * DMODEL * sizeof(__bf16);   // 8 MiB
  const size_t wBf     = (size_t)DMODEL * DMODEL * sizeof(__bf16);  // 2 MiB
  const size_t projF32 = (size_t)NROWS * DMODEL * sizeof(float);    // 16 MiB
  char* ws = (char*)d_ws;
  __bf16* in_bf[5];
  for (int i = 0; i < 5; ++i) in_bf[i] = (__bf16*)(ws + i * actBf);
  // aliases into the (dead after projections) activation-bf16 region:
  __bf16* q_b   = (__bf16*)(ws + 0 * actBf);
  __bf16* k_b   = (__bf16*)(ws + 1 * actBf);
  __bf16* vT_b  = (__bf16*)(ws + 2 * actBf);
  __bf16* ctx_b = (__bf16*)(ws + 3 * actBf);
  __bf16* w_bf[6];
  for (int i = 0; i < 6; ++i) w_bf[i] = (__bf16*)(ws + 5 * actBf + i * wBf);
  float* proj[5];
  for (int i = 0; i < 5; ++i)
    proj[i] = (float*)(ws + 5 * actBf + 6 * wBf + i * projF32);

  dim3 blk(256);

  // 0) precision conversion (done once; L2-resident afterwards)
  const int actBlocks = (NROWS * DMODEL) / (256 * 8);               // 2048
  for (int i = 0; i < 5; ++i)
    convert_bf16_kernel<<<dim3(actBlocks), blk, 0, stream>>>(act[i], in_bf[i]);
  for (int i = 0; i < 6; ++i)
    transpose_convert_kernel<<<dim3(32, 32), blk, 0, stream>>>(Wlin[i], w_bf[i]);

  // 1) five input projections (pure-bf16 WMMA GEMM, double-buffered)
  dim3 gGrid(NROWS / 128, DMODEL / 128);                            // 32 x 8
  for (int i = 0; i < 5; ++i)
    gemm_bf16_kernel<<<gGrid, blk, 0, stream>>>(in_bf[i], w_bf[i], blin[i],
                                                proj[i], NROWS, DMODEL, DMODEL);

  // 2) gating, mixing, pack to attention layouts (proj order: qx qc kx kc vx)
  gate_pack_kernel<<<dim3(NROWS), blk, 0, stream>>>(
      proj[0], proj[3], proj[1], proj[4], proj[2],
      W_Vqx, b_Vqx, W_Vqc, b_Vqc, W_Vkx, b_Vkx, W_Vkc, b_Vkc,
      q_b, k_b, vT_b);

  // 3) flash attention per (b,h)
  dim3 aGrid(S_LEN / 128, BATCH * NHEAD);                           // 16 x 32
  flash_attn_kernel<<<aGrid, blk, 0, stream>>>(q_b, k_b, vT_b, ctx_b);

  // 4) output projection -> [S,B,D] d_out (row m = s*B+b is identity here)
  gemm_bf16_kernel<<<gGrid, blk, 0, stream>>>(ctx_b, w_bf[5], blin[5],
                                              (float*)d_out, NROWS, DMODEL, DMODEL);
}
